// SparseAutoencoderBase_47296179864058
// MI455X (gfx1250) — compile-verified, matrix operand order checked by reading, not run on hardware
//
#include <hip/hip_runtime.h>
#include <stdint.h>

// Problem constants (match reference)
#define B_ 4096
#define D_ 2048
#define L_ 16384
#define K_ 64

typedef __attribute__((ext_vector_type(16))) __bf16 v16bf;
typedef __attribute__((ext_vector_type(8)))  float  v8f;
typedef __attribute__((ext_vector_type(8)))  unsigned short ushort8;
typedef __attribute__((ext_vector_type(4)))  int    v4i;

#define AS1 __attribute__((address_space(1)))
#define AS3 __attribute__((address_space(3)))

#if defined(__has_builtin)
#if __has_builtin(__builtin_amdgcn_global_load_async_to_lds_b128) && \
    __has_builtin(__builtin_amdgcn_s_wait_asynccnt)
#define USE_ASYNC_LDS 1
#endif
#endif
#ifndef USE_ASYNC_LDS
#define USE_ASYNC_LDS 0
#endif

union FragU { v16bf v; ushort8 h[2]; };

__device__ __forceinline__ unsigned short f2bf(float f) {
  unsigned u = __float_as_uint(f);
  unsigned r = u + 0x7fffu + ((u >> 16) & 1u);   // round-to-nearest-even
  return (unsigned short)(r >> 16);
}
__device__ __forceinline__ float bf2f(unsigned short s) {
  return __uint_as_float(((unsigned)s) << 16);
}

__device__ __forceinline__ AS1 v4i* as_global_v4i(const unsigned short* p) {
  return (AS1 v4i*)(void*)const_cast<unsigned short*>(p);
}
__device__ __forceinline__ AS3 v4i* as_lds_v4i(unsigned short* p) {
  return (AS3 v4i*)p;
}

// ---------------------------------------------------------------------------
// Kernel 0a: convert x (fp32) -> bf16
// ---------------------------------------------------------------------------
__global__ void cvt_x_kernel(const float* __restrict__ x,
                             unsigned short* __restrict__ xb, int n) {
  int i = blockIdx.x * blockDim.x + threadIdx.x;
  int stride = gridDim.x * blockDim.x;
  for (; i < n; i += stride) xb[i] = f2bf(x[i]);
}

// ---------------------------------------------------------------------------
// Kernel 0b: build W^T in bf16: W [D,L] row-major -> WT [L,D] row-major.
// ---------------------------------------------------------------------------
__global__ void build_wt_kernel(const float* __restrict__ W,
                                unsigned short* __restrict__ WT) {
  long i = (long)blockIdx.x * blockDim.x + threadIdx.x;
  const long n = (long)D_ * L_;
  const long stride = (long)gridDim.x * blockDim.x;
  for (; i < n; i += stride) {
    const int d = (int)(i / L_);
    const int l = (int)(i % L_);
    WT[(long)l * D_ + d] = f2bf(W[i]);
  }
}

// ---------------------------------------------------------------------------
// Kernel 1: encode GEMM  z = x @ W + b_enc   via v_wmma_f32_16x16x32_bf16.
// Block tile 128x256, K-step 32. 8 waves (2x4), each wave a 64x64 tile.
// Async global->LDS double-buffered staging (ASYNCcnt path).
// B-fragments are loaded one at a time to keep the live VGPR set under 256.
// ---------------------------------------------------------------------------
#define BM 128
#define BN 256
#define BK 32
#define NK (D_ / BK)
#define LDST 40  // bf16 elems per LDS row (32 + 8 pad -> conflict-free frags)

__global__ __launch_bounds__(256) void encode_kernel(
    const unsigned short* __restrict__ xb,   // [B_, D_] bf16 bits
    const unsigned short* __restrict__ wt,   // [L_, D_] bf16 bits (W^T)
    const float* __restrict__ b_enc,         // [L_]
    float* __restrict__ z)                   // [B_, L_]
{
  const int tid  = threadIdx.x;
  const int lane = tid & 31;
  const int wid  = tid >> 5;
  const int wm   = wid >> 2;   // 0..1
  const int wn   = wid & 3;    // 0..3

  const int m0 = blockIdx.y * BM;
  const int n0 = blockIdx.x * BN;

  v8f acc[4][4];
  const v8f vzero = {0.f, 0.f, 0.f, 0.f, 0.f, 0.f, 0.f, 0.f};
#pragma unroll
  for (int i = 0; i < 4; ++i)
#pragma unroll
    for (int j = 0; j < 4; ++j) acc[i][j] = vzero;

  // cooperative load mapping
  const int a_row = tid >> 1;           // 0..127
  const int a_off = (tid & 1) * 16;     // 0 or 16 (bf16 elems)
  const int b_row = tid;                // 0..255

  const int laneHalf = lane >> 4;       // 0/1
  const int laneMod  = lane & 15;
  const int k0 = laneHalf * 8;          // fragment K sub-run start

#if USE_ASYNC_LDS
  // ---- async double-buffered staging (no VGPR bounce, ASYNCcnt tracked) ----
  __shared__ unsigned short Alds[2][BM * LDST];
  __shared__ unsigned short Blds[2][BN * LDST];

  const unsigned short* gA0 = xb + (long)(m0 + a_row) * D_ + a_off;
  const unsigned short* gB0 = wt + (long)(n0 + b_row) * D_;

  // prologue: stage k=0 into buffer 0 (6 async b128 per thread: 2 A + 4 B)
  {
    AS1 v4i* ga = as_global_v4i(gA0);
    AS1 v4i* gb = as_global_v4i(gB0);
    AS3 v4i* la = as_lds_v4i(&Alds[0][a_row * LDST + a_off]);
    AS3 v4i* lb = as_lds_v4i(&Blds[0][b_row * LDST]);
    __builtin_amdgcn_global_load_async_to_lds_b128(ga, la, 0, 0);
    __builtin_amdgcn_global_load_async_to_lds_b128(ga, la, 16, 0);
    __builtin_amdgcn_global_load_async_to_lds_b128(gb, lb, 0, 0);
    __builtin_amdgcn_global_load_async_to_lds_b128(gb, lb, 16, 0);
    __builtin_amdgcn_global_load_async_to_lds_b128(gb, lb, 32, 0);
    __builtin_amdgcn_global_load_async_to_lds_b128(gb, lb, 48, 0);
  }

  for (int ki = 0; ki < NK; ++ki) {
    const int cur = ki & 1;
    if (ki + 1 < NK) {
      // stage next K-step into the other buffer; safe: the trailing barrier
      // of iteration ki-1 guaranteed all waves finished reading it.
      const int kn = (ki + 1) * BK;
      AS1 v4i* ga = as_global_v4i(gA0 + kn);
      AS1 v4i* gb = as_global_v4i(gB0 + kn);
      AS3 v4i* la = as_lds_v4i(&Alds[cur ^ 1][a_row * LDST + a_off]);
      AS3 v4i* lb = as_lds_v4i(&Blds[cur ^ 1][b_row * LDST]);
      __builtin_amdgcn_global_load_async_to_lds_b128(ga, la, 0, 0);
      __builtin_amdgcn_global_load_async_to_lds_b128(ga, la, 16, 0);
      __builtin_amdgcn_global_load_async_to_lds_b128(gb, lb, 0, 0);
      __builtin_amdgcn_global_load_async_to_lds_b128(gb, lb, 16, 0);
      __builtin_amdgcn_global_load_async_to_lds_b128(gb, lb, 32, 0);
      __builtin_amdgcn_global_load_async_to_lds_b128(gb, lb, 48, 0);
      // async loads complete in order: <=6 outstanding means cur's 6 landed
      __builtin_amdgcn_s_wait_asynccnt(6);
    } else {
      __builtin_amdgcn_s_wait_asynccnt(0);
    }
    __syncthreads();  // all waves' data for `cur` visible in LDS

    // A fragments stay resident; B fragments streamed one at a time so the
    // live set (128 acc + 32 A + 8 B) stays under the 256-VGPR window.
    FragU a[4];
#pragma unroll
    for (int mi = 0; mi < 4; ++mi) {
      const int r = wm * 64 + mi * 16 + laneMod;
      a[mi].h[0] = *(const ushort8*)&Alds[cur][r * LDST + k0];
      a[mi].h[1] = *(const ushort8*)&Alds[cur][r * LDST + k0 + 16];
    }
#pragma unroll
    for (int ni = 0; ni < 4; ++ni) {
      const int r = wn * 64 + ni * 16 + laneMod;
      FragU b;
      b.h[0] = *(const ushort8*)&Blds[cur][r * LDST + k0];
      b.h[1] = *(const ushort8*)&Blds[cur][r * LDST + k0 + 16];
#pragma unroll
      for (int mi = 0; mi < 4; ++mi)
        acc[mi][ni] = __builtin_amdgcn_wmma_f32_16x16x32_bf16(
            false, a[mi].v, false, b.v, (short)0, acc[mi][ni],
            false, false);
    }

    __syncthreads();  // done reading `cur`; next iter may overwrite it
  }
#else
  // ---- fallback: synchronous VGPR-bounce staging (proven path) ----
  __shared__ unsigned short Alds[BM * LDST];
  __shared__ unsigned short Blds[BN * LDST];

  for (int k = 0; k < D_; k += BK) {
    const ushort8* ga =
        (const ushort8*)(xb + (long)(m0 + a_row) * D_ + k + a_off);
    ushort8 av0 = ga[0];
    ushort8 av1 = ga[1];
    const ushort8* gb = (const ushort8*)(wt + (long)(n0 + b_row) * D_ + k);
    ushort8 bv0 = gb[0], bv1 = gb[1], bv2 = gb[2], bv3 = gb[3];

    __syncthreads();
    *(ushort8*)&Alds[a_row * LDST + a_off]     = av0;
    *(ushort8*)&Alds[a_row * LDST + a_off + 8] = av1;
    *(ushort8*)&Blds[b_row * LDST +  0] = bv0;
    *(ushort8*)&Blds[b_row * LDST +  8] = bv1;
    *(ushort8*)&Blds[b_row * LDST + 16] = bv2;
    *(ushort8*)&Blds[b_row * LDST + 24] = bv3;
    __syncthreads();

    FragU a[4];
#pragma unroll
    for (int mi = 0; mi < 4; ++mi) {
      const int r = wm * 64 + mi * 16 + laneMod;
      a[mi].h[0] = *(const ushort8*)&Alds[r * LDST + k0];
      a[mi].h[1] = *(const ushort8*)&Alds[r * LDST + k0 + 16];
    }
#pragma unroll
    for (int ni = 0; ni < 4; ++ni) {
      const int r = wn * 64 + ni * 16 + laneMod;
      FragU b;
      b.h[0] = *(const ushort8*)&Blds[r * LDST + k0];
      b.h[1] = *(const ushort8*)&Blds[r * LDST + k0 + 16];
#pragma unroll
      for (int mi = 0; mi < 4; ++mi)
        acc[mi][ni] = __builtin_amdgcn_wmma_f32_16x16x32_bf16(
            false, a[mi].v, false, b.v, (short)0, acc[mi][ni],
            false, false);
    }
  }
#endif

  // epilogue: add b_enc, store dense z (masked in-place by topk kernel later)
#pragma unroll
  for (int mi = 0; mi < 4; ++mi) {
    const int rbase = m0 + wm * 64 + mi * 16 + laneHalf * 8;
#pragma unroll
    for (int ni = 0; ni < 4; ++ni) {
      const int col = n0 + wn * 64 + ni * 16 + laneMod;
      const float be = b_enc[col];
#pragma unroll
      for (int r = 0; r < 8; ++r)
        z[(long)(rbase + r) * L_ + col] = acc[mi][ni][r] + be;
    }
  }
}

// ---------------------------------------------------------------------------
// Kernel 2: per-row exact top-K select on key = bits(|z|) (monotonic for
// non-negative floats). 4x 8-bit radix passes w/ LDS histograms, then mask
// z in place and emit compact (idx,val) lists for the sparse decoder.
// ---------------------------------------------------------------------------
__global__ __launch_bounds__(256) void topk_kernel(
    float* __restrict__ z,           // [B_, L_] in/out (z -> z_sparse)
    float* __restrict__ nz_val,      // [B_, K_]
    int*   __restrict__ nz_idx)      // [B_, K_]
{
  __shared__ unsigned hist[256];
  __shared__ unsigned s_byte;
  __shared__ unsigned s_cum;
  __shared__ unsigned s_tie;
  __shared__ unsigned s_out;

  const int row = blockIdx.x;
  const int tid = threadIdx.x;
  float* zr = z + (long)row * L_;

  unsigned prefix = 0;
  int remaining = K_;

  for (int pass = 0; pass < 4; ++pass) {
    const int shift = 24 - 8 * pass;
    hist[tid] = 0;
    __syncthreads();
    const unsigned hmask = (pass == 0) ? 0u : (0xFFFFFFFFu << (shift + 8));
    for (int i = tid; i < L_; i += 256) {
      const unsigned key = __float_as_uint(zr[i]) & 0x7FFFFFFFu;
      if ((key & hmask) == (prefix & hmask))
        atomicAdd(&hist[(key >> shift) & 255u], 1u);
    }
    __syncthreads();
    if (tid == 0) {
      unsigned cum = 0;
      int b = 255;
      for (; b > 0; --b) {
        if (cum + hist[b] >= (unsigned)remaining) break;
        cum += hist[b];
      }
      s_byte = (unsigned)b;
      s_cum  = cum;
    }
    __syncthreads();
    prefix |= (s_byte << shift);
    remaining -= (int)s_cum;
    __syncthreads();
  }

  // prefix == K-th largest key T. keep key>T; first `remaining` ties at ==T.
  if (tid == 0) { s_tie = 0; s_out = 0; }
  __syncthreads();
  for (int i = tid; i < L_; i += 256) {
    const float v = zr[i];
    const unsigned key = __float_as_uint(v) & 0x7FFFFFFFu;
    bool keep = false;
    if (key > prefix) {
      keep = true;
    } else if (key == prefix) {
      const unsigned p = atomicAdd(&s_tie, 1u);
      keep = (p < (unsigned)remaining);
    }
    if (keep) {
      const unsigned slot = atomicAdd(&s_out, 1u);
      nz_val[row * K_ + slot] = v;
      nz_idx[row * K_ + slot] = i;
    } else {
      zr[i] = 0.0f;
    }
  }
}

// ---------------------------------------------------------------------------
// Kernel 3: sparse decode  recon[row,:] = b_dec + sum_j val_j * WT[idx_j,:].
// WT rows are contiguous bf16 (64MB total -> L2-resident on 192MB L2).
// ---------------------------------------------------------------------------
__global__ __launch_bounds__(256) void decode_kernel(
    const float* __restrict__ nz_val,
    const int*   __restrict__ nz_idx,
    const unsigned short* __restrict__ wt,   // [L_, D_] bf16 bits
    const float* __restrict__ b_dec,         // [D_]
    float* __restrict__ recon)               // [B_, D_]
{
  __shared__ float s_val[K_];
  __shared__ int   s_idx[K_];
  const int row = blockIdx.x;
  const int tid = threadIdx.x;
  if (tid < K_) {
    s_val[tid] = nz_val[row * K_ + tid];
    s_idx[tid] = nz_idx[row * K_ + tid];
  }
  __syncthreads();

  const int d0 = tid * 8;                     // 256*8 == D_
  float acc[8];
#pragma unroll
  for (int j = 0; j < 8; ++j) acc[j] = b_dec[d0 + j];

  for (int t = 0; t < K_; ++t) {
    const float v = s_val[t];
    const ushort8 w = *(const ushort8*)&wt[(long)s_idx[t] * D_ + d0];
#pragma unroll
    for (int j = 0; j < 8; ++j) acc[j] += v * bf2f(w[j]);
  }

  float* out = recon + (long)row * D_ + d0;
#pragma unroll
  for (int j = 0; j < 8; ++j) out[j] = acc[j];
}

// ---------------------------------------------------------------------------
extern "C" void kernel_launch(void* const* d_in, const int* in_sizes, int n_in,
                              void* d_out, int out_size, void* d_ws,
                              size_t ws_size, hipStream_t stream) {
  const float* x     = (const float*)d_in[0];   // [B_, D_]
  const float* W_enc = (const float*)d_in[1];   // [D_, L_]
  const float* b_enc = (const float*)d_in[2];   // [L_]
  const float* b_dec = (const float*)d_in[3];   // [D_]
  // d_in[4] == k (64), baked in as K_

  float* recon = (float*)d_out;                       // [B_, D_]
  float* z     = (float*)d_out + (long)B_ * D_;       // [B_, L_] (z_sparse)

  char* ws = (char*)d_ws;
  unsigned short* xb = (unsigned short*)ws;  ws += (size_t)B_ * D_ * 2;  // 16MB
  unsigned short* wt = (unsigned short*)ws;  ws += (size_t)L_ * D_ * 2;  // 64MB
  float* nz_val = (float*)ws;                ws += (size_t)B_ * K_ * 4;  // 1MB
  int*   nz_idx = (int*)ws;                                              // 1MB

  cvt_x_kernel<<<2048, 256, 0, stream>>>(x, xb, B_ * D_);
  build_wt_kernel<<<8192, 256, 0, stream>>>(W_enc, wt);

  dim3 grid(L_ / BN, B_ / BM);  // (64, 32)
  encode_kernel<<<grid, 256, 0, stream>>>(xb, wt, b_enc, z);

  topk_kernel<<<B_, 256, 0, stream>>>(z, nz_val, nz_idx);
  decode_kernel<<<B_, 256, 0, stream>>>(nz_val, nz_idx, wt, b_dec, recon);
}
